// MemoryGAT_87608742903868
// MI455X (gfx1250) — matrix operand order repaired and checked by reading, not compiled, not run on hardware
//
#include <hip/hip_runtime.h>
#include <hip/hip_bf16.h>

typedef __attribute__((ext_vector_type(16))) __bf16 v16bf;
typedef __attribute__((ext_vector_type(8)))  float  v8f;
typedef int v4i_pkt __attribute__((vector_size(16)));   // packet type for async-LDS builtin

#define NNODES 100000
#define NEDGES 400000

#if defined(__has_builtin)
#  if __has_builtin(__builtin_amdgcn_global_load_async_to_lds_b128)
#    define USE_ASYNC_LDS 1
#  endif
#endif

__device__ __forceinline__ void wait_async_lds() {
#if defined(__has_builtin) && __has_builtin(__builtin_amdgcn_s_wait_asynccnt)
    __builtin_amdgcn_s_wait_asynccnt(0);
#else
    asm volatile("s_wait_asynccnt 0" ::: "memory");
#endif
}

__device__ __forceinline__ float wave_sum(float v) {
    #pragma unroll
    for (int o = 16; o > 0; o >>= 1) v += __shfl_xor(v, o, 32);
    return v;
}

__device__ __forceinline__ float gelu_exact(float x) {
    return 0.5f * x * (1.f + erff(x * 0.70710678118654752f));
}

// ---------------------------------------------------------------------------
// Fill zeros (accumulator init; must run every launch for determinism)
// ---------------------------------------------------------------------------
__global__ void fill0_kernel(float* p, size_t n) {
    size_t i  = (size_t)blockIdx.x * blockDim.x + threadIdx.x;
    size_t st = (size_t)gridDim.x * blockDim.x;
    for (; i < n; i += st) p[i] = 0.f;
}

// ---------------------------------------------------------------------------
// Input projection: h0 = LN(GELU(concat(x, ctx) @ Wp + bp))  [N,64]
// ---------------------------------------------------------------------------
__global__ void proj_kernel(const float* __restrict__ x, const float* __restrict__ ctx,
                            const float* __restrict__ Wp, const float* __restrict__ bp,
                            const float* __restrict__ g, const float* __restrict__ b,
                            float* __restrict__ h0, int n_nodes) {
    __shared__ float sctx[256];
    int tid = threadIdx.x;
    if (tid < 256) sctx[tid] = ctx[tid];
    __syncthreads();
    int wave = tid >> 5, lane = tid & 31;
    int n = blockIdx.x * 8 + wave;
    if (n >= n_nodes) return;
    const float* xr = x + (size_t)n * 267;
    float a0 = 0.f, a1 = 0.f;
    for (int j = 0; j < 267; j++) {
        float xv = xr[j];
        a0 += xv * Wp[j * 64 + lane];
        a1 += xv * Wp[j * 64 + lane + 32];
    }
    for (int j = 0; j < 256; j++) {
        float xv = sctx[j];
        int jj = 267 + j;
        a0 += xv * Wp[jj * 64 + lane];
        a1 += xv * Wp[jj * 64 + lane + 32];
    }
    a0 = gelu_exact(a0 + bp[lane]);
    a1 = gelu_exact(a1 + bp[lane + 32]);
    float s  = wave_sum(a0 + a1);
    float s2 = wave_sum(a0 * a0 + a1 * a1);
    float m   = s * (1.f / 64.f);
    float inv = rsqrtf(s2 * (1.f / 64.f) - m * m + 1e-5f);
    h0[(size_t)n * 64 + lane]      = (a0 - m) * inv * g[lane] + b[lane];
    h0[(size_t)n * 64 + lane + 32] = (a1 - m) * inv * g[lane + 32] + b[lane + 32];
}

// ---------------------------------------------------------------------------
// WMMA GEMM: Y[M,No] = X[M,K] @ W[K,No] (+bias), fp32 in/out, bf16 compute.
// 8 waves/block, each wave owns a 16x64 strip (4 16x16x32 WMMA tiles).
// ---------------------------------------------------------------------------
__global__ void gemm_wmma_kernel(const float* __restrict__ X, const float* __restrict__ W,
                                 const float* __restrict__ bias, float* __restrict__ Y,
                                 int M, int K, int Nout) {
    __shared__ __align__(16) __bf16 sB[4 * 32 * 16];   // 4 col tiles, swizzled
    int tid = threadIdx.x;
    int wave = tid >> 5, lane = tid & 31;
    int tile16 = blockIdx.x * 8 + wave;
    bool active = (tile16 * 16 + 15) < M;
    int r0 = tile16 * 16;
    int c0 = blockIdx.y * 64;
    int r = lane & 15, half = lane >> 4;
    v8f acc[4] = {v8f{}, v8f{}, v8f{}, v8f{}};

    // hoist per-lane bias values (no branches in the store loop)
    float bv[4] = {0.f, 0.f, 0.f, 0.f};
    if (bias != nullptr) {
        #pragma unroll
        for (int t = 0; t < 4; t++) bv[t] = bias[c0 + t * 16 + (lane & 15)];
    }

    for (int k0 = 0; k0 < K; k0 += 32) {
        __syncthreads();
        // cooperative stage of W[k0:k0+32, c0:c0+64] -> swizzled bf16 LDS (float4 loads)
        for (int e4 = tid; e4 < 512; e4 += 256) {
            int e = e4 * 4;
            int k = e >> 6, nn = e & 63;          // vec4 never crosses a K-row
            float4 v4 = *reinterpret_cast<const float4*>(W + (size_t)(k0 + k) * Nout + c0 + nn);
            float vals[4] = {v4.x, v4.y, v4.z, v4.w};
            int Lk = (((k >> 3) & 1) << 4);
            int kk = (k & 7) | ((k >> 4) << 3);
            #pragma unroll
            for (int q = 0; q < 4; q++) {
                int n = nn + q;
                int t = n >> 4, n16 = n & 15;
                sB[((t * 32 + Lk + n16) << 4) + kk] = (__bf16)vals[q];
            }
        }
        if (k0 + 32 < K)
            __builtin_prefetch(W + (size_t)(k0 + 32) * Nout + c0 + (tid & 63), 0, 1);
        __syncthreads();
        if (active) {
            const float* xp = X + (size_t)(r0 + r) * K + k0 + half * 8;
            float4 f0 = *reinterpret_cast<const float4*>(xp);
            float4 f1 = *reinterpret_cast<const float4*>(xp + 4);
            float4 f2 = *reinterpret_cast<const float4*>(xp + 16);
            float4 f3 = *reinterpret_cast<const float4*>(xp + 20);
            v16bf a;
            a[0]=(__bf16)f0.x; a[1]=(__bf16)f0.y; a[2]=(__bf16)f0.z; a[3]=(__bf16)f0.w;
            a[4]=(__bf16)f1.x; a[5]=(__bf16)f1.y; a[6]=(__bf16)f1.z; a[7]=(__bf16)f1.w;
            a[8]=(__bf16)f2.x; a[9]=(__bf16)f2.y; a[10]=(__bf16)f2.z; a[11]=(__bf16)f2.w;
            a[12]=(__bf16)f3.x; a[13]=(__bf16)f3.y; a[14]=(__bf16)f3.z; a[15]=(__bf16)f3.w;
            #pragma unroll
            for (int t = 0; t < 4; t++) {
                v16bf bfrag = *reinterpret_cast<const v16bf*>(&sB[(t * 32 + lane) << 4]);
                acc[t] = __builtin_amdgcn_wmma_f32_16x16x32_bf16(
                    false, a, false, bfrag, (short)0, acc[t], false, false);
            }
        }
    }
    if (active) {
        #pragma unroll
        for (int t = 0; t < 4; t++) {
            #pragma unroll
            for (int j = 0; j < 8; j++) {
                int row = r0 + j + half * 8;
                int col = c0 + t * 16 + (lane & 15);
                Y[(size_t)row * Nout + col] = acc[t][j] + bv[t];
            }
        }
    }
}

// ---------------------------------------------------------------------------
// Edge kernel 1: alpha[e,h] = sum_d lrelu(hs[s]+ht[t]+eattr@We)[h,d] * A[h,d]
// We/A staged global->LDS with async-to-LDS DMA (s_wait_asynccnt) when
// available; wave per edge, 8 dims/lane, head = lane>>3.
// ---------------------------------------------------------------------------
__global__ void edge_alpha_kernel(const float* __restrict__ hs, const float* __restrict__ ht,
                                  const float* __restrict__ eattr, const float* __restrict__ We,
                                  const float* __restrict__ A,
                                  const int* __restrict__ src, const int* __restrict__ tgt,
                                  float* __restrict__ alpha, unsigned int* __restrict__ amax,
                                  int nE) {
    __shared__ __align__(16) float sWe[11 * 256];
    __shared__ __align__(16) float sA[256];
    int tid = threadIdx.x;
#ifdef USE_ASYNC_LDS
    for (int i = tid; i < 704; i += 256)          // 11*256 floats = 704 x b128
        __builtin_amdgcn_global_load_async_to_lds_b128(
            (v4i_pkt*)(We + i * 4), (v4i_pkt*)(sWe + i * 4), 0, 0);
    if (tid < 64)                                  // 256 floats = 64 x b128
        __builtin_amdgcn_global_load_async_to_lds_b128(
            (v4i_pkt*)(A + tid * 4), (v4i_pkt*)(sA + tid * 4), 0, 0);
    wait_async_lds();
#else
    for (int i = tid; i < 11 * 256; i += 256) sWe[i] = We[i];
    if (tid < 256) sA[tid] = A[tid];
#endif
    __syncthreads();
    int wave = tid >> 5, lane = tid & 31;
    int e = blockIdx.x * 8 + wave;
    if (e >= nE) return;
    int s = src[e], t = tgt[e];
    int d = lane * 8;
    float4 a0 = *reinterpret_cast<const float4*>(hs + (size_t)s * 256 + d);
    float4 a1 = *reinterpret_cast<const float4*>(hs + (size_t)s * 256 + d + 4);
    float4 b0 = *reinterpret_cast<const float4*>(ht + (size_t)t * 256 + d);
    float4 b1 = *reinterpret_cast<const float4*>(ht + (size_t)t * 256 + d + 4);
    float m[8] = {a0.x + b0.x, a0.y + b0.y, a0.z + b0.z, a0.w + b0.w,
                  a1.x + b1.x, a1.y + b1.y, a1.z + b1.z, a1.w + b1.w};
    const float* ea = eattr + (size_t)e * 11;
    for (int j = 0; j < 11; j++) {
        float ev = ea[j];
        const float* wj = &sWe[j * 256 + d];
        #pragma unroll
        for (int i = 0; i < 8; i++) m[i] += ev * wj[i];
    }
    int h = lane >> 3;
    float p = 0.f;
    #pragma unroll
    for (int i = 0; i < 8; i++) {
        float mm = m[i] > 0.f ? m[i] : 0.2f * m[i];
        p += mm * sA[h * 64 + (d & 63) + i];
    }
    p += __shfl_xor(p, 1, 32);
    p += __shfl_xor(p, 2, 32);
    p += __shfl_xor(p, 4, 32);
    if ((lane & 7) == 0) {
        alpha[(size_t)e * 4 + h] = p;
        atomicMax(&amax[(size_t)t * 4 + h], __float_as_uint(fmaxf(p, 0.f)));
    }
}

// ---------------------------------------------------------------------------
// Edge kernel 2: a = exp(alpha - amax[t]); segment-sum into asum
// ---------------------------------------------------------------------------
__global__ void edge_exp_kernel(const float* __restrict__ alpha, const unsigned int* __restrict__ amax,
                                const int* __restrict__ tgt, float* __restrict__ aexp,
                                float* __restrict__ asum, int nE) {
    int i = blockIdx.x * blockDim.x + threadIdx.x;
    if (i >= nE * 4) return;
    int e = i >> 2, h = i & 3;
    int t = tgt[e];
    float am = __uint_as_float(amax[(size_t)t * 4 + h]);
    float a = __expf(alpha[i] - am);
    aexp[i] = a;
    atomicAdd(&asum[(size_t)t * 4 + h], a);
}

// ---------------------------------------------------------------------------
// Edge kernel 3: gat[t] += hs[s] * a/(asum[t]+1e-8)   (f32 atomic scatter)
// ---------------------------------------------------------------------------
__global__ void edge_scatter_kernel(const float* __restrict__ hs, const float* __restrict__ aexp,
                                    const float* __restrict__ asum,
                                    const int* __restrict__ src, const int* __restrict__ tgt,
                                    float* __restrict__ gat, int nE) {
    int wave = threadIdx.x >> 5, lane = threadIdx.x & 31;
    int e = blockIdx.x * 8 + wave;
    if (e >= nE) return;
    int s = src[e], t = tgt[e];
    int h = lane >> 3, d = lane * 8;
    float w = aexp[(size_t)e * 4 + h] / (asum[(size_t)t * 4 + h] + 1e-8f);
    const float* hp = hs + (size_t)s * 256 + d;
    float* gp = gat + (size_t)t * 256 + d;
    #pragma unroll
    for (int i = 0; i < 8; i++) atomicAdd(&gp[i], hp[i] * w);
}

// ---------------------------------------------------------------------------
// Layer epilogue (concat layers): out = GELU(LN(gat + res))   C = 256
// ---------------------------------------------------------------------------
__global__ void node_epilogue256_kernel(const float* __restrict__ gat, const float* __restrict__ res,
                                        const float* __restrict__ g, const float* __restrict__ b,
                                        float* __restrict__ out, int n_nodes) {
    int wave = threadIdx.x >> 5, lane = threadIdx.x & 31;
    int n = blockIdx.x * 8 + wave;
    if (n >= n_nodes) return;
    const float* gr = gat + (size_t)n * 256 + lane * 8;
    const float* rr = res + (size_t)n * 256 + lane * 8;
    float v[8], s = 0.f, s2 = 0.f;
    #pragma unroll
    for (int i = 0; i < 8; i++) { v[i] = gr[i] + rr[i]; s += v[i]; s2 += v[i] * v[i]; }
    s = wave_sum(s); s2 = wave_sum(s2);
    float m   = s * (1.f / 256.f);
    float inv = rsqrtf(s2 * (1.f / 256.f) - m * m + 1e-5f);
    float* o = out + (size_t)n * 256 + lane * 8;
    #pragma unroll
    for (int i = 0; i < 8; i++) {
        int c = lane * 8 + i;
        o[i] = gelu_exact((v[i] - m) * inv * g[c] + b[c]);
    }
}

// ---------------------------------------------------------------------------
// Layer-2 epilogue: head-mean then GELU(LN(mean + res))   C = 64
// ---------------------------------------------------------------------------
__global__ void node_epilogue64_kernel(const float* __restrict__ gat, const float* __restrict__ res,
                                       const float* __restrict__ g, const float* __restrict__ b,
                                       float* __restrict__ out, int n_nodes) {
    int wave = threadIdx.x >> 5, lane = threadIdx.x & 31;
    int n = blockIdx.x * 8 + wave;
    if (n >= n_nodes) return;
    const float* gr = gat + (size_t)n * 256;
    int l2 = lane + 32;
    float v0 = 0.25f * (gr[lane] + gr[64 + lane] + gr[128 + lane] + gr[192 + lane])
             + res[(size_t)n * 64 + lane];
    float v1 = 0.25f * (gr[l2] + gr[64 + l2] + gr[128 + l2] + gr[192 + l2])
             + res[(size_t)n * 64 + l2];
    float s  = wave_sum(v0 + v1);
    float s2 = wave_sum(v0 * v0 + v1 * v1);
    float m   = s * (1.f / 64.f);
    float inv = rsqrtf(s2 * (1.f / 64.f) - m * m + 1e-5f);
    out[(size_t)n * 64 + lane] = gelu_exact((v0 - m) * inv * g[lane] + b[lane]);
    out[(size_t)n * 64 + l2]   = gelu_exact((v1 - m) * inv * g[l2] + b[l2]);
}

// ---------------------------------------------------------------------------
// Score head: sigmoid(GELU(h@Wh1+bh1)@Wh2+bh2)   64 -> 32 -> 1, wave per node
// ---------------------------------------------------------------------------
__global__ void head_kernel(const float* __restrict__ h, const float* __restrict__ Wh1,
                            const float* __restrict__ bh1, const float* __restrict__ Wh2,
                            const float* __restrict__ bh2, float* __restrict__ out, int n_nodes) {
    int wave = threadIdx.x >> 5, lane = threadIdx.x & 31;
    int n = blockIdx.x * 8 + wave;
    if (n >= n_nodes) return;
    const float* hr = h + (size_t)n * 64;
    float z = 0.f;
    for (int j = 0; j < 64; j++) z += hr[j] * Wh1[j * 32 + lane];
    z = gelu_exact(z + bh1[lane]);
    float p = wave_sum(z * Wh2[lane]);
    if (lane == 0) out[n] = 1.f / (1.f + __expf(-(p + bh2[0])));
}

// ---------------------------------------------------------------------------
extern "C" void kernel_launch(void* const* d_in, const int* in_sizes, int n_in,
                              void* d_out, int out_size, void* d_ws, size_t ws_size,
                              hipStream_t stream) {
    (void)in_sizes; (void)n_in; (void)out_size; (void)ws_size;
    const int N = NNODES, E = NEDGES;

    const float* x    = (const float*)d_in[0];
    const float* ctx  = (const float*)d_in[1];
    const float* eatt = (const float*)d_in[2];
    const float* Wp   = (const float*)d_in[3];
    const float* bp   = (const float*)d_in[4];
    const float* g_in = (const float*)d_in[5];
    const float* b_in = (const float*)d_in[6];
    const float* Ws0  = (const float*)d_in[7];
    const float* Wt0  = (const float*)d_in[8];
    const float* A0   = (const float*)d_in[9];
    const float* We0  = (const float*)d_in[10];
    const float* Sk0W = (const float*)d_in[11];
    const float* Sk0b = (const float*)d_in[12];
    const float* gn0  = (const float*)d_in[13];
    const float* bn0  = (const float*)d_in[14];
    const float* Ws1  = (const float*)d_in[15];
    const float* Wt1  = (const float*)d_in[16];
    const float* A1   = (const float*)d_in[17];
    const float* We1  = (const float*)d_in[18];
    const float* gn1  = (const float*)d_in[19];
    const float* bn1  = (const float*)d_in[20];
    const float* Ws2  = (const float*)d_in[21];
    const float* Wt2  = (const float*)d_in[22];
    const float* A2   = (const float*)d_in[23];
    const float* We2  = (const float*)d_in[24];
    const float* Sk2W = (const float*)d_in[25];
    const float* Sk2b = (const float*)d_in[26];
    const float* gn2  = (const float*)d_in[27];
    const float* bn2  = (const float*)d_in[28];
    const float* Wh1  = (const float*)d_in[29];
    const float* bh1  = (const float*)d_in[30];
    const float* Wh2  = (const float*)d_in[31];
    const float* bh2  = (const float*)d_in[32];
    const int*   eidx = (const int*)d_in[33];
    const int* src = eidx;          // edge_index[0]
    const int* tgt = eidx + E;      // edge_index[1]
    float* out = (float*)d_out;

    // workspace partition
    float* w = (float*)d_ws;
    float* h0   = w; w += (size_t)N * 64;
    float* hA   = w; w += (size_t)N * 256;   // h after layer 0
    float* hB   = w; w += (size_t)N * 256;   // h after layer 1
    float* hsB  = w; w += (size_t)N * 256;
    float* htB  = w; w += (size_t)N * 256;
    float* resB = w; w += (size_t)N * 256;
    float* gat  = w; w += (size_t)N * 256;
    float* h3   = w; w += (size_t)N * 64;
    float* alpha= w; w += (size_t)E * 4;
    float* aexp = w; w += (size_t)E * 4;
    float* amax = w; w += (size_t)N * 4;
    float* asum = w; w += (size_t)N * 4;

    const int NB8 = (N + 7) / 8;      // wave-per-node grids
    const int EB8 = (E + 7) / 8;      // wave-per-edge grids

    auto gemm = [&](const float* X, const float* W, const float* bias, float* Y,
                    int M, int K, int Nout) {
        dim3 grid((((M + 15) / 16) + 7) / 8, Nout / 64);
        gemm_wmma_kernel<<<grid, 256, 0, stream>>>(X, W, bias, Y, M, K, Nout);
    };
    auto edge_phase = [&](const float* We, const float* A) {
        fill0_kernel<<<2048, 256, 0, stream>>>(gat, (size_t)N * 256);
        fill0_kernel<<<256, 256, 0, stream>>>(amax, (size_t)N * 4);
        fill0_kernel<<<256, 256, 0, stream>>>(asum, (size_t)N * 4);
        edge_alpha_kernel<<<EB8, 256, 0, stream>>>(hsB, htB, eatt, We, A, src, tgt,
                                                   alpha, (unsigned int*)amax, E);
        edge_exp_kernel<<<(E * 4 + 255) / 256, 256, 0, stream>>>(alpha, (const unsigned int*)amax,
                                                                 tgt, aexp, asum, E);
        edge_scatter_kernel<<<EB8, 256, 0, stream>>>(hsB, aexp, asum, src, tgt, gat, E);
    };

    // input projection
    proj_kernel<<<NB8, 256, 0, stream>>>(x, ctx, Wp, bp, g_in, b_in, h0, N);

    // layer 0: 64 -> 256 concat, projected skip
    gemm(h0, Ws0, nullptr, hsB, N, 64, 256);
    gemm(h0, Wt0, nullptr, htB, N, 64, 256);
    gemm(h0, Sk0W, Sk0b, resB, N, 64, 256);
    edge_phase(We0, A0);
    node_epilogue256_kernel<<<NB8, 256, 0, stream>>>(gat, resB, gn0, bn0, hA, N);

    // layer 1: 256 -> 256 concat, identity skip
    gemm(hA, Ws1, nullptr, hsB, N, 256, 256);
    gemm(hA, Wt1, nullptr, htB, N, 256, 256);
    edge_phase(We1, A1);
    node_epilogue256_kernel<<<NB8, 256, 0, stream>>>(gat, hA, gn1, bn1, hB, N);

    // layer 2: 256 -> 64 head-mean, projected skip
    gemm(hB, Ws2, nullptr, hsB, N, 256, 256);
    gemm(hB, Wt2, nullptr, htB, N, 256, 256);
    gemm(hB, Sk2W, Sk2b, resB, N, 256, 64);
    edge_phase(We2, A2);
    node_epilogue64_kernel<<<NB8, 256, 0, stream>>>(gat, resB, gn2, bn2, h3, N);

    // score head
    head_kernel<<<NB8, 256, 0, stream>>>(h3, Wh1, bh1, Wh2, bh2, out, N);
}
